// HybridBlock_10479720202317
// MI455X (gfx1250) — matrix operand order repaired and checked by reading, not compiled, not run on hardware
//
#include <hip/hip_runtime.h>
#include <hip/hip_bf16.h>
#include <cstdint>

// ---------------------------------------------------------------------------
// Problem constants (match reference)
// ---------------------------------------------------------------------------
#define Bc   4
#define Tc   2048
#define Cc   1024
#define Hc   16
#define Dc   64
#define FFNc 4096
#define Mrows (Bc * Tc)          // 8192
#define EPSc 1e-6f

typedef __attribute__((ext_vector_type(16))) __bf16 v16bf;
typedef __attribute__((ext_vector_type(8)))  float  v8f;

// ---- gfx1250 async global->LDS path (guarded; sync fallback) --------------
#ifndef __has_builtin
#define __has_builtin(x) 0
#endif
#if __has_builtin(__builtin_amdgcn_global_load_async_to_lds_b128)
#define HAVE_ASYNC_LDS 1
#else
#define HAVE_ASYNC_LDS 0
#endif

#if HAVE_ASYNC_LDS
typedef __attribute__((__vector_size__(16))) int v4i_;
typedef __attribute__((address_space(1))) v4i_ glb_v4i;
typedef __attribute__((address_space(3))) v4i_ lds_v4i;
__device__ __forceinline__ void async_copy_b128(void* lds, const void* glb) {
    // generic LDS pointer low 32 bits == LDS offset (ISA 10.2 aperture rules);
    // generic global pointer value == AS1 value.
    glb_v4i* g = (glb_v4i*)(uintptr_t)glb;
    lds_v4i* l = (lds_v4i*)(unsigned int)(uintptr_t)lds;
    __builtin_amdgcn_global_load_async_to_lds_b128(g, l, 0, 0);
}
__device__ __forceinline__ void wait_async0() {
    asm volatile("s_wait_asynccnt 0x0" ::: "memory");
}
#endif

__device__ __forceinline__ float bf2f(unsigned short u) {
    union { unsigned int i; float f; } c;
    c.i = ((unsigned int)u) << 16;
    return c.f;
}
__device__ __forceinline__ unsigned short f2bf(float f) {
    union { float f; unsigned int i; } c;
    c.f = f;
    unsigned int r = c.i + 0x7FFFu + ((c.i >> 16) & 1u);   // round-nearest-even
    return (unsigned short)(r >> 16);
}
__device__ __forceinline__ float sigmoidf_(float x) {
    return 1.0f / (1.0f + __expf(-x));
}

// ---------------------------------------------------------------------------
// Weight fp32 -> bf16 cast
// ---------------------------------------------------------------------------
__global__ void cast_bf16_kernel(const float* __restrict__ in,
                                 unsigned short* __restrict__ out, int n) {
    int i = blockIdx.x * blockDim.x + threadIdx.x;
    if (i < n) out[i] = f2bf(in[i]);
}

// ---------------------------------------------------------------------------
// RMSNorm (one block per token row, C = 1024, 256 thr x 4 elems) -> bf16
// ---------------------------------------------------------------------------
__global__ __launch_bounds__(256)
void rmsnorm_bf16_kernel(const float* __restrict__ x,
                         const float* __restrict__ w,
                         unsigned short* __restrict__ out) {
    __shared__ float red[8];
    const int row = blockIdx.x;
    const int tid = threadIdx.x;
    const int lane = tid & 31, wid = tid >> 5;

    const float4 xv = ((const float4*)(x + (size_t)row * Cc))[tid];
    float s = xv.x * xv.x + xv.y * xv.y + xv.z * xv.z + xv.w * xv.w;
    #pragma unroll
    for (int o = 16; o > 0; o >>= 1) s += __shfl_xor(s, o, 32);
    if (lane == 0) red[wid] = s;
    __syncthreads();
    float tot = 0.f;
    #pragma unroll
    for (int i = 0; i < 8; ++i) tot += red[i];
    const float inv = rsqrtf(tot * (1.0f / Cc) + EPSc);

    const float4 wv = ((const float4*)w)[tid];
    ushort4 o4;
    o4.x = f2bf(xv.x * inv * wv.x);
    o4.y = f2bf(xv.y * inv * wv.y);
    o4.z = f2bf(xv.z * inv * wv.z);
    o4.w = f2bf(xv.w * inv * wv.w);
    *(ushort4*)(out + (size_t)row * Cc + tid * 4) = o4;
}

// ---------------------------------------------------------------------------
// bf16 GEMM:  out[M,N] = A[M,K](bf16) * W[N,K](bf16)^T   (fp32 accumulate)
// Block tile 128x128, 8 waves, wave tile 64x32 (4x2 WMMA 16x16x32_bf16).
// Double-buffered LDS; async global->LDS when available (ASYNCcnt tracked),
// one barrier per K-step, DMA overlapped with WMMA.
// Epilogues: 0 f32 store | 1 sigmoid f32 | 2 silu->bf16 | 3 acc*other->bf16
//            4 acc+residual -> f32
// ---------------------------------------------------------------------------
#define BM 128
#define BN 128
#define BK 32

union Frag { v16bf v; uint4 u[2]; };

template <int EPI>
__global__ __launch_bounds__(256)
void gemm_bf16_kernel(const unsigned short* __restrict__ A,
                      const unsigned short* __restrict__ W,
                      int N, int K,
                      float* __restrict__ outF,
                      unsigned short* __restrict__ outB,
                      const float* __restrict__ residual,
                      const unsigned short* __restrict__ other) {
    __shared__ __align__(16) unsigned short As[2][BM * BK];
    __shared__ __align__(16) unsigned short Bs[2][BN * BK];

    const int tid  = threadIdx.x;
    const int lane = tid & 31;
    const int wave = tid >> 5;
    const int wr   = wave >> 2;      // 0..1  (M direction, 64 rows each)
    const int wc   = wave & 3;       // 0..3  (N direction, 32 cols each)
    const int bm   = blockIdx.y * BM;
    const int bn   = blockIdx.x * BN;

    const int l15   = lane & 15;
    const int khalf = lane >> 4;

    v8f acc[4][2];
    #pragma unroll
    for (int i = 0; i < 4; ++i)
        #pragma unroll
        for (int j = 0; j < 2; ++j)
            acc[i][j] = (v8f)(0.0f);

    const int lrow = tid >> 1;            // 0..127
    const int lcol = (tid & 1) * 16;      // 0 or 16  (ushorts)

    const unsigned short* gA = A + (size_t)(bm + lrow) * K + lcol;
    const unsigned short* gW = W + (size_t)(bn + lrow) * K + lcol;
    unsigned short* sA = &As[0][0] + lrow * BK + lcol;
    unsigned short* sB = &Bs[0][0] + lrow * BK + lcol;

    auto stage = [&](int ib, int kk) {
        unsigned short* dA = sA + ib * (BM * BK);
        unsigned short* dB = sB + ib * (BN * BK);
#if HAVE_ASYNC_LDS
        async_copy_b128(dA,     gA + kk);
        async_copy_b128(dA + 8, gA + kk + 8);
        async_copy_b128(dB,     gW + kk);
        async_copy_b128(dB + 8, gW + kk + 8);
#else
        uint4 a0 = *(const uint4*)(gA + kk);
        uint4 a1 = *(const uint4*)(gA + kk + 8);
        uint4 w0 = *(const uint4*)(gW + kk);
        uint4 w1 = *(const uint4*)(gW + kk + 8);
        *(uint4*)(dA)     = a0;
        *(uint4*)(dA + 8) = a1;
        *(uint4*)(dB)     = w0;
        *(uint4*)(dB + 8) = w1;
#endif
    };

    const int nt = K / BK;
    stage(0, 0);
#if HAVE_ASYNC_LDS
    wait_async0();
#endif
    __syncthreads();

    for (int it = 0; it < nt; ++it) {
        const int ib = it & 1;
        if (it + 1 < nt) stage(ib ^ 1, (it + 1) * BK);

        const unsigned short* Ab = &As[ib][0];
        const unsigned short* Bb = &Bs[ib][0];

        Frag a[4], b[2];
        #pragma unroll
        for (int j = 0; j < 2; ++j) {
            const int n = wc * 32 + j * 16 + l15;
            b[j].u[0] = *(const uint4*)(Bb + n * BK + khalf * 16);
            b[j].u[1] = *(const uint4*)(Bb + n * BK + khalf * 16 + 8);
        }
        #pragma unroll
        for (int i = 0; i < 4; ++i) {
            const int r = wr * 64 + i * 16 + l15;
            a[i].u[0] = *(const uint4*)(Ab + r * BK + khalf * 16);       // K 0..7 / 16..23
            a[i].u[1] = *(const uint4*)(Ab + r * BK + khalf * 16 + 8);   // K 8..15 / 24..31
        }
        #pragma unroll
        for (int i = 0; i < 4; ++i)
            #pragma unroll
            for (int j = 0; j < 2; ++j)
                acc[i][j] = __builtin_amdgcn_wmma_f32_16x16x32_bf16(
                    false, a[i].v, false, b[j].v, (short)0, acc[i][j], false, false);

#if HAVE_ASYNC_LDS
        wait_async0();        // tile it+1 landed (overlapped with WMMAs above)
#endif
        __syncthreads();      // all waves done reading buf ib / writing ib^1
    }

    // C/D layout: VGPR r -> M = r (lanes 0-15) / 8+r (lanes 16-31); N = lane&15
    const int mbase = bm + wr * 64;
    const int nbase = bn + wc * 32;
    #pragma unroll
    for (int i = 0; i < 4; ++i)
        #pragma unroll
        for (int j = 0; j < 2; ++j)
            #pragma unroll
            for (int r = 0; r < 8; ++r) {
                const int m = mbase + i * 16 + khalf * 8 + r;
                const int n = nbase + j * 16 + l15;
                const size_t idx = (size_t)m * N + n;
                const float v = acc[i][j][r];
                if constexpr (EPI == 0) outF[idx] = v;
                if constexpr (EPI == 1) outF[idx] = sigmoidf_(v);
                if constexpr (EPI == 2) outB[idx] = f2bf(v * sigmoidf_(v));
                if constexpr (EPI == 3) outB[idx] = f2bf(v * bf2f(other[idx]));
                if constexpr (EPI == 4) outF[idx] = v + residual[idx];
            }
}

// ---------------------------------------------------------------------------
// beta = sigmoid(xn @ beta_w^T + beta_b)    (N = 16, tiny)
// ---------------------------------------------------------------------------
__global__ void beta_kernel(const unsigned short* __restrict__ xn,
                            const float* __restrict__ bw,
                            const float* __restrict__ bb,
                            float* __restrict__ beta) {
    const int idx = blockIdx.x * blockDim.x + threadIdx.x;
    if (idx >= Mrows * Hc) return;
    const int row = idx / Hc;
    const int h   = idx - row * Hc;
    const unsigned short* xr = xn + (size_t)row * Cc;
    const float* wr = bw + (size_t)h * Cc;
    float s = 0.f;
    #pragma unroll 8
    for (int k = 0; k < Cc; ++k) s += bf2f(xr[k]) * wr[k];
    beta[idx] = sigmoidf_(s + bb[h]);
}

// ---------------------------------------------------------------------------
// Gated delta-rule scan.  One block per (b,h); 64 threads; thread d owns
// state row M[d][0..63] in VGPRs.  q,k broadcast via LDS.  Serial in T.
// ---------------------------------------------------------------------------
__global__ __launch_bounds__(64)
void delta_scan_kernel(const float* __restrict__ q,
                       const float* __restrict__ k,
                       const float* __restrict__ v,
                       const float* __restrict__ beta,
                       float* __restrict__ o) {
    const int bh = blockIdx.x;
    const int b  = bh >> 4;       // / Hc
    const int h  = bh & 15;
    const int d  = threadIdx.x;
    const int lane = d & 31, wid = d >> 5;

    __shared__ float q_s[Dc], k_s[Dc], red[4];

    float Mr[Dc];
    #pragma unroll
    for (int c = 0; c < Dc; ++c) Mr[c] = 0.f;

    const size_t base = ((size_t)b * Tc) * Cc + (size_t)h * Dc;

    for (int t = 0; t < Tc; ++t) {
        const size_t off = base + (size_t)t * Cc;
        const float qv = q[off + d];
        const float kv = k[off + d];
        const float vv = v[off + d];
        const float bt = beta[((size_t)b * Tc + t) * Hc + h];

        float s1 = qv * qv, s2 = kv * kv;
        #pragma unroll
        for (int m = 16; m > 0; m >>= 1) {
            s1 += __shfl_xor(s1, m, 32);
            s2 += __shfl_xor(s2, m, 32);
        }
        if (lane == 0) { red[wid] = s1; red[2 + wid] = s2; }
        __syncthreads();
        const float qn = qv / fmaxf(sqrtf(red[0] + red[1]), 1e-12f);
        const float kn = kv / fmaxf(sqrtf(red[2] + red[3]), 1e-12f);
        q_s[d] = qn;
        k_s[d] = kn;
        __syncthreads();

        float ov = 0.f, dv = vv;
        #pragma unroll 16
        for (int c = 0; c < Dc; ++c) {
            ov = fmaf(Mr[c], q_s[c], ov);
            dv = fmaf(-Mr[c], k_s[c], dv);
        }
        const float bd = bt * dv;
        #pragma unroll 16
        for (int c = 0; c < Dc; ++c) Mr[c] = fmaf(bd, k_s[c], Mr[c]);

        o[off + d] = ov;
    }
}

// ---------------------------------------------------------------------------
// gated_bf16 = bf16(gate_sigmoid * o)
// ---------------------------------------------------------------------------
__global__ void gated_kernel(const float* __restrict__ gate,
                             const float* __restrict__ o,
                             unsigned short* __restrict__ out, int n) {
    int i = blockIdx.x * blockDim.x + threadIdx.x;
    if (i < n) out[i] = f2bf(gate[i] * o[i]);
}

// ---------------------------------------------------------------------------
// Launcher
// ---------------------------------------------------------------------------
extern "C" void kernel_launch(void* const* d_in, const int* in_sizes, int n_in,
                              void* d_out, int out_size, void* d_ws, size_t ws_size,
                              hipStream_t stream) {
    (void)in_sizes; (void)n_in; (void)out_size; (void)ws_size;

    const float* x      = (const float*)d_in[0];
    // d_in[1] cos, d_in[2] sin: unused (GatedDeltaNet branch)
    const float* n1w    = (const float*)d_in[3];
    const float* n2w    = (const float*)d_in[4];
    const float* qw     = (const float*)d_in[5];
    const float* kw     = (const float*)d_in[6];
    const float* vw     = (const float*)d_in[7];
    const float* betaw  = (const float*)d_in[8];
    const float* betab  = (const float*)d_in[9];
    const float* gatew  = (const float*)d_in[10];
    const float* ow     = (const float*)d_in[11];
    const float* fgw    = (const float*)d_in[12];
    const float* fuw    = (const float*)d_in[13];
    const float* fdw    = (const float*)d_in[14];
    float* out          = (float*)d_out;

    // ---- workspace carve-out -------------------------------------------
    uint8_t* base = (uint8_t*)d_ws;
    size_t off = 0;
    auto alloc = [&](size_t bytes) -> void* {
        void* r = base + off;
        off += (bytes + 255) & ~(size_t)255;
        return r;
    };
    const size_t nCC = (size_t)Cc * Cc;       // 1M
    const size_t nFC = (size_t)FFNc * Cc;     // 4M
    unsigned short* wq_b  = (unsigned short*)alloc(nCC * 2);
    unsigned short* wk_b  = (unsigned short*)alloc(nCC * 2);
    unsigned short* wv_b  = (unsigned short*)alloc(nCC * 2);
    unsigned short* wg_b  = (unsigned short*)alloc(nCC * 2);
    unsigned short* wo_b  = (unsigned short*)alloc(nCC * 2);
    unsigned short* wfg_b = (unsigned short*)alloc(nFC * 2);
    unsigned short* wfu_b = (unsigned short*)alloc(nFC * 2);
    unsigned short* wfd_b = (unsigned short*)alloc(nFC * 2);
    unsigned short* xn1_b = (unsigned short*)alloc((size_t)Mrows * Cc * 2);
    unsigned short* xn2_b = (unsigned short*)alloc((size_t)Mrows * Cc * 2);
    float* qbuf    = (float*)alloc((size_t)Mrows * Cc * 4);
    float* kbuf    = (float*)alloc((size_t)Mrows * Cc * 4);
    float* vbuf    = (float*)alloc((size_t)Mrows * Cc * 4);
    float* gatebuf = (float*)alloc((size_t)Mrows * Cc * 4);
    float* betabuf = (float*)alloc((size_t)Mrows * Hc * 4);
    float* obuf    = (float*)alloc((size_t)Mrows * Cc * 4);
    unsigned short* gated_b = (unsigned short*)alloc((size_t)Mrows * Cc * 2);
    unsigned short* g_b     = (unsigned short*)alloc((size_t)Mrows * FFNc * 2);
    unsigned short* h_b     = (unsigned short*)alloc((size_t)Mrows * FFNc * 2);

    // ---- 1) weights -> bf16 --------------------------------------------
    auto cast = [&](const float* src, unsigned short* dst, size_t n) {
        cast_bf16_kernel<<<(unsigned)((n + 255) / 256), 256, 0, stream>>>(src, dst, (int)n);
    };
    cast(qw,  wq_b,  nCC);  cast(kw,  wk_b,  nCC);  cast(vw, wv_b, nCC);
    cast(gatew, wg_b, nCC); cast(ow,  wo_b,  nCC);
    cast(fgw, wfg_b, nFC);  cast(fuw, wfu_b, nFC);  cast(fdw, wfd_b, nFC);

    // ---- 2) xn1 = rmsnorm(x) -> bf16 -----------------------------------
    rmsnorm_bf16_kernel<<<Mrows, 256, 0, stream>>>(x, n1w, xn1_b);

    // ---- 3) projections -------------------------------------------------
    dim3 blk(256);
    dim3 gCC(Cc / BN, Mrows / BM);          // N=1024
    dim3 gFC(FFNc / BN, Mrows / BM);        // N=4096
    gemm_bf16_kernel<0><<<gCC, blk, 0, stream>>>(xn1_b, wq_b, Cc, Cc, qbuf, nullptr, nullptr, nullptr);
    gemm_bf16_kernel<0><<<gCC, blk, 0, stream>>>(xn1_b, wk_b, Cc, Cc, kbuf, nullptr, nullptr, nullptr);
    gemm_bf16_kernel<0><<<gCC, blk, 0, stream>>>(xn1_b, wv_b, Cc, Cc, vbuf, nullptr, nullptr, nullptr);
    gemm_bf16_kernel<1><<<gCC, blk, 0, stream>>>(xn1_b, wg_b, Cc, Cc, gatebuf, nullptr, nullptr, nullptr);

    // ---- 4) beta --------------------------------------------------------
    beta_kernel<<<(Mrows * Hc + 255) / 256, 256, 0, stream>>>(xn1_b, betaw, betab, betabuf);

    // ---- 5) delta-rule scan --------------------------------------------
    delta_scan_kernel<<<Bc * Hc, Dc, 0, stream>>>(qbuf, kbuf, vbuf, betabuf, obuf);

    // ---- 6) gated = sigmoid(gate) * o -> bf16 --------------------------
    gated_kernel<<<(Mrows * Cc + 255) / 256, 256, 0, stream>>>(gatebuf, obuf, gated_b, Mrows * Cc);

    // ---- 7) x1 = x + gated @ o_w^T  (into d_out) -----------------------
    gemm_bf16_kernel<4><<<gCC, blk, 0, stream>>>(gated_b, wo_b, Cc, Cc, out, nullptr, x, nullptr);

    // ---- 8) xn2 = rmsnorm(x1) -> bf16 ----------------------------------
    rmsnorm_bf16_kernel<<<Mrows, 256, 0, stream>>>(out, n2w, xn2_b);

    // ---- 9) FFN ---------------------------------------------------------
    gemm_bf16_kernel<2><<<gFC, blk, 0, stream>>>(xn2_b, wfg_b, FFNc, Cc, nullptr, g_b, nullptr, nullptr);
    gemm_bf16_kernel<3><<<gFC, blk, 0, stream>>>(xn2_b, wfu_b, FFNc, Cc, nullptr, h_b, nullptr, g_b);
    gemm_bf16_kernel<4><<<gCC, blk, 0, stream>>>(h_b, wfd_b, Cc, FFNc, out, nullptr, out, nullptr);
}
